// CondConv2D_60627758350912
// MI455X (gfx1250) — matrix operand order repaired and checked by reading, not compiled, not run on hardware
//
#include <hip/hip_runtime.h>
#include <hip/hip_bf16.h>

// ---------- types ----------
typedef __attribute__((ext_vector_type(16))) __bf16 bf16x16;
typedef __attribute__((ext_vector_type(8)))  __bf16 bf16x8;
typedef __attribute__((ext_vector_type(8)))  float  f32x8;
typedef __attribute__((ext_vector_type(4)))  float  f32x4;
typedef __attribute__((ext_vector_type(4)))  unsigned int su32x4;
typedef __attribute__((ext_vector_type(8)))  unsigned int su32x8;

union ABfrag { bf16x16 v16; bf16x8 v8[2]; };
union Pack4  { unsigned long long u; __bf16 p[4]; };

// Problem constants (from reference)
#define NB   32
#define HH   128
#define WW   128
#define CIN  64
#define FF   128
#define EE   4
#define CD   64
#define KK   576                    // 3*3*64
#define KSTEPS 18                   // 576/32
#define BSLICE (KSTEPS * FF * 32)   // 73728 bf16 elems per sample
#define FHALF  64                   // F columns per block
#define BSUB   (KSTEPS * FHALF * 32) // 36864 bf16 elems per block slice

// Workspace byte offsets
#define WS_ALPHAS 0         // 32*4 f32
#define WS_BIAS   4096      // 32*128 f32
#define WS_BMIX   65536     // 32*73728 bf16 = 4.72 MB

// ---------------- kernel 1: routing softmax ----------------
__global__ void routing_kernel(const float* __restrict__ cond,
                               const float* __restrict__ aw,
                               const float* __restrict__ ab,
                               float* __restrict__ alphas) {
    int b = threadIdx.x;           // 32 threads, one wave
    float lg[EE];
    #pragma unroll
    for (int e = 0; e < EE; ++e) lg[e] = ab[e];
    for (int c = 0; c < CD; ++c) {
        float cv = cond[b * CD + c];
        #pragma unroll
        for (int e = 0; e < EE; ++e) lg[e] += cv * aw[c * EE + e];
    }
    float mx = lg[0];
    #pragma unroll
    for (int e = 1; e < EE; ++e) mx = fmaxf(mx, lg[e]);
    float sum = 0.f;
    #pragma unroll
    for (int e = 0; e < EE; ++e) { lg[e] = __expf(lg[e] - mx); sum += lg[e]; }
    float inv = 1.f / sum;
    #pragma unroll
    for (int e = 0; e < EE; ++e) alphas[b * EE + e] = lg[e] * inv;
}

// ---------------- kernel 2a: mixed bias ----------------
__global__ void bias_mix_kernel(const float* __restrict__ alphas,
                                const float* __restrict__ ebias,
                                float* __restrict__ bias) {
    int idx = blockIdx.x * 256 + threadIdx.x;      // < 32*128
    int b = idx >> 7, f = idx & 127;
    float v = 0.f;
    #pragma unroll
    for (int e = 0; e < EE; ++e) v += alphas[b * EE + e] * ebias[e * FF + f];
    bias[idx] = v;
}

// ---------------- kernel 2b: mixed kernels, B-fragment-swizzled bf16 ----------------
// Bmix layout: [b][s(18)][n(128)][ko(32)], k = s*32+ko, value = sum_e a[b][e]*EK[e][k][n]
__global__ void kmix_kernel(const float* __restrict__ EK,
                            const float* __restrict__ alphas,
                            __bf16* __restrict__ Bmix) {
    int idx = blockIdx.x * 256 + threadIdx.x;      // < 32*73728 exact
    int b    = idx / BSLICE;
    int rem  = idx - b * BSLICE;
    int s    = rem / (FF * 32);
    int rem2 = rem - s * (FF * 32);
    int n    = rem2 >> 5;
    int ko   = rem2 & 31;
    int k    = s * 32 + ko;
    float a0 = alphas[b * EE + 0], a1 = alphas[b * EE + 1];
    float a2 = alphas[b * EE + 2], a3 = alphas[b * EE + 3];
    const float* ek = EK + (size_t)k * FF + n;
    const size_t estr = (size_t)KK * FF;
    float v = a0 * ek[0] + a1 * ek[estr] + a2 * ek[2 * estr] + a3 * ek[3 * estr];
    Bmix[idx] = (__bf16)v;
}

// ---------------- fragment load helpers ----------------
__device__ __forceinline__ ABfrag load_afrag(const __bf16* xs, int kh, int xw,
                                             int ci0, int lhalf) {
    const int abase = (kh * 130 + xw) * CIN + ci0 + lhalf * 16;
    ABfrag A;
    A.v8[0] = *(const bf16x8*)&xs[abase];        // K ci0+{0..7 | 16..23}
    A.v8[1] = *(const bf16x8*)&xs[abase + 8];    // K ci0+{8..15 | 24..31}
    return A;
}
__device__ __forceinline__ ABfrag load_bfrag(const __bf16* Bs, int s, int t,
                                             int l15, int lhalf) {
    const int bbase = (s * FHALF + (t * 16 + l15)) * 32 + lhalf * 16;
    ABfrag B;
    B.v8[0] = *(const bf16x8*)&Bs[bbase];
    B.v8[1] = *(const bf16x8*)&Bs[bbase + 8];
    return B;
}

// ---------------- kernel 3: implicit-GEMM conv via bf16 WMMA ----------------
// One block = (sample b, output row h, F-half fh). 256 threads = 8 waves.
// LDS: Bs [18][64][32] bf16 (73728 B, TDM-staged, strided 2D copy)
//    + xs [3][130][64] bf16 (49920 B)  => 123648 B -> 2 workgroups per WGP
__global__ void __launch_bounds__(256)
condconv_main(const float* __restrict__ x,
              const __bf16* __restrict__ Bmix,
              const float* __restrict__ bias,
              float* __restrict__ out) {
    extern __shared__ char smem[];
    __bf16* Bs = (__bf16*)smem;                        // LDS offset 0
    __bf16* xs = (__bf16*)(smem + BSUB * 2);           // LDS offset 73728

    const int bi  = blockIdx.x;        // [b(32)][h(128)][fh(2)]
    const int b   = bi >> 8;
    const int h   = (bi >> 1) & 127;
    const int fh  = bi & 1;
    const int tid = threadIdx.x;

    // ---- stage this block's mixed-kernel slice via Tensor Data Mover ----
    // Strided 2D copy: 18 rows of 2048 bf16 elems, row stride 4096 elems.
    if (tid < 32) {   // wave 0 issues the DMA; TDM ignores EXEC
        unsigned long long gaddr = (unsigned long long)(const void*)
            (Bmix + (size_t)b * BSLICE + fh * (FHALF * 32));
        su32x4 g0;
        g0[0] = 1u;                                        // count=1, user D#
        g0[1] = 0u;                                        // lds_addr = 0 (Bs base)
        g0[2] = (unsigned)gaddr;                           // global_addr[31:0]
        g0[3] = ((unsigned)(gaddr >> 32) & 0x01FFFFFFu)    // global_addr[56:32]
                | (2u << 30);                              // type = 2 (image)
        su32x8 g1;
        g1[0] = 1u << 16;        // workgroup_mask=0, data_size=1 (2 bytes)
        g1[1] = 2048u << 16;     // tensor_dim0[15:0]=2048 (bits [63:48])
        g1[2] = 18u << 16;       // tensor_dim0[31:16]=0, tensor_dim1[15:0]=18
        g1[3] = 2048u << 16;     // tensor_dim1[31:16]=0, tile_dim0=2048
        g1[4] = 18u;             // tile_dim1=18, tile_dim2=0
        g1[5] = 4096u;           // tensor_dim0_stride[31:0]=4096 elems
        g1[6] = 0u;              // stride0[47:32]=0, tensor_dim1_stride[15:0]=0
        g1[7] = 0u;
        asm volatile("tensor_load_to_lds %0, %1" :: "s"(g0), "s"(g1) : "memory");
        __builtin_amdgcn_s_wait_tensorcnt(0);
    }

    // ---- zero the W-halo columns (w=0 and w=129) ----
    if (tid < 96) {
        int r = tid / 32, rem = tid & 31;
        int wp = (rem >> 4) ? 129 : 0;
        int cq = rem & 15;
        *(unsigned long long*)&xs[(r * 130 + wp) * CIN + cq * 4] = 0ULL;
    }

    // ---- stage x rows h-1..h+1 as bf16 (zero pad out-of-range rows) ----
    for (int c = tid; c < 3 * WW * (CIN / 4); c += 256) {
        int r   = c / (WW * 16);
        int rem = c - r * (WW * 16);
        int wq  = rem >> 4;
        int cq  = rem & 15;
        int h2  = h + r - 1;
        f32x4 v = {0.f, 0.f, 0.f, 0.f};
        if ((unsigned)h2 < (unsigned)HH)
            v = *(const f32x4*)&x[((((size_t)b * HH + h2) * WW + wq) * CIN) + cq * 4];
        Pack4 pk;
        pk.p[0] = (__bf16)v[0]; pk.p[1] = (__bf16)v[1];
        pk.p[2] = (__bf16)v[2]; pk.p[3] = (__bf16)v[3];
        *(unsigned long long*)&xs[(r * 130 + (wq + 1)) * CIN + cq * 4] = pk.u;
    }

    __syncthreads();

    const int wave  = tid >> 5;
    const int lane  = tid & 31;
    const int lhalf = lane >> 4;      // 0: K-low half, 1: K-high half
    const int l15   = lane & 15;
    const int wtile = wave * 16;      // this wave's 16 output w-positions

    f32x8 acc[4];
    #pragma unroll
    for (int t = 0; t < 4; ++t) acc[t] = (f32x8){0.f,0.f,0.f,0.f,0.f,0.f,0.f,0.f};

    // Software-pipelined: A(s+1) and B(s,t+1) loads issued before WMMA(s,t)
    ABfrag A_cur = load_afrag(xs, 0, wtile + l15 + 0, 0, lhalf);  // s=0

    #pragma unroll
    for (int s = 0; s < KSTEPS; ++s) {
        ABfrag A_next = A_cur;
        ABfrag Bf[2];
        Bf[0] = load_bfrag(Bs, s, 0, l15, lhalf);

        #pragma unroll
        for (int t = 0; t < 4; ++t) {
            if (t < 3) {
                Bf[(t + 1) & 1] = load_bfrag(Bs, s, t + 1, l15, lhalf);
            } else if (s < KSTEPS - 1) {
                const int sn  = s + 1;
                const int tap = sn >> 1;
                const int kh  = tap / 3;
                const int kw  = tap - kh * 3;
                A_next = load_afrag(xs, kh, wtile + l15 + kw, (sn & 1) * 32, lhalf);
            }
            acc[t] = __builtin_amdgcn_wmma_f32_16x16x32_bf16(
                false, A_cur.v16, false, Bf[t & 1].v16, (short)0, acc[t],
                false, false);
        }
        A_cur = A_next;
    }

    // ---- epilogue: + bias, store fp32 ----
    float* outp = out + (((size_t)b * HH + h) * WW) * FF;
    #pragma unroll
    for (int t = 0; t < 4; ++t) {
        const int f  = fh * FHALF + t * 16 + l15;
        const float bv = bias[b * FF + f];
        #pragma unroll
        for (int j = 0; j < 8; ++j) {
            const int m = j + lhalf * 8;                   // C/D layout row
            outp[(size_t)(wtile + m) * FF + f] = acc[t][j] + bv;
        }
    }
}

// ---------------- host launcher ----------------
extern "C" void kernel_launch(void* const* d_in, const int* in_sizes, int n_in,
                              void* d_out, int out_size, void* d_ws, size_t ws_size,
                              hipStream_t stream) {
    const float* x      = (const float*)d_in[0];
    const float* cond   = (const float*)d_in[1];
    const float* aw     = (const float*)d_in[2];
    const float* ab     = (const float*)d_in[3];
    const float* ek     = (const float*)d_in[4];
    const float* ebias  = (const float*)d_in[5];
    float* out = (float*)d_out;

    char* ws = (char*)d_ws;
    float*  alphas = (float*)(ws + WS_ALPHAS);
    float*  bias   = (float*)(ws + WS_BIAS);
    __bf16* Bmix   = (__bf16*)(ws + WS_BMIX);

    routing_kernel<<<1, 32, 0, stream>>>(cond, aw, ab, alphas);
    bias_mix_kernel<<<(NB * FF) / 256, 256, 0, stream>>>(alphas, ebias, bias);
    kmix_kernel<<<(NB * BSLICE) / 256, 256, 0, stream>>>(ek, alphas, Bmix);

    const size_t lds_bytes = (size_t)(BSUB + 3 * 130 * CIN) * 2;  // 123648 B
    condconv_main<<<NB * HH * 2, 256, lds_bytes, stream>>>(x, Bmix, bias, out);
}
